// CompositionalMLP_75763223101514
// MI455X (gfx1250) — matrix operand order repaired and checked by reading, not compiled, not run on hardware
//
#include <hip/hip_runtime.h>

// ---------------------------------------------------------------------------
// CompositionalMLP (MoE-style block-diagonal 2-layer MLP) for MI455X / gfx1250
//
// Roofline: 32.2 GFLOP vs ~272 MB HBM traffic -> AI ~119 FLOP/B. Need the
// 16-bit WMMA path (v_wmma_f32_16x16x32_f16, f32 accumulate) to approach the
// 23.3 TB/s memory floor; f32 WMMA (K=4) would be hopelessly compute bound.
// Weights are converted f32->f16 once into d_ws (~2 MB) and then live in the
// 192 MB L2 across all 4096 workgroups.
//
// Epilogue tricks:
//  - bias is pre-loaded into the WMMA accumulator (C/D layout: column == lane,
//    so all 8 acc VGPRs share one per-lane bias) -> no adds in the epilogue.
//  - leaky(v) == max(v, 0.2*v) for slope < 1 -> v_mul + v_max_num_f32, no
//    v_cmp/v_cndmask chains.
// ---------------------------------------------------------------------------

typedef _Float16 half8 __attribute__((ext_vector_type(8)));
typedef _Float16 half4 __attribute__((ext_vector_type(4)));
typedef _Float16 v16h  __attribute__((ext_vector_type(16)));
typedef float    v8f   __attribute__((ext_vector_type(8)));

#define KEXP 16      // experts
#define M    128     // per-expert input width
#define DK   240     // hidden width
#define DKP  256     // hidden width padded to a multiple of 32 (WMMA K-step)
#define L    128     // per-expert output width
#define BT   64      // batch rows per workgroup
#define NCOL (KEXP * M)   // 2048 = row stride of x and out
#define SLOPE 0.2f

// Concatenate two 8-half fragments into the 16-half WMMA operand register set.
__device__ __forceinline__ v16h cat16(half8 lo, half8 hi) {
    return __builtin_shufflevector(lo, hi, 0,1,2,3,4,5,6,7,8,9,10,11,12,13,14,15);
}

// slope < 1  =>  leaky(v) == max(v, slope*v). Lowers to mul + v_max_num_f32.
__device__ __forceinline__ float leaky(float v) {
    return __builtin_fmaxf(v, SLOPE * v);
}

// ---------------------------------------------------------------------------
// Pre-pass 1: extract expert-diagonal block of W1 and convert to f16.
// w1c layout: [k][d][m]  (K-dim m contiguous -> B-fragment rows contiguous)
// ---------------------------------------------------------------------------
__global__ void convert_w1_kernel(const float* __restrict__ W1,
                                  _Float16* __restrict__ w1c) {
    int o = blockIdx.x * blockDim.x + threadIdx.x;
    if (o >= KEXP * DK * M) return;
    int k = o / (DK * M);
    int r = o % (DK * M);
    int d = r / M;
    int m = r % M;
    // W1 is [K, DK, K*M]; only the (k,k) diagonal block is ever used.
    float v = W1[(size_t)(k * DK + d) * NCOL + (size_t)k * M + m];
    w1c[o] = (_Float16)v;
}

// ---------------------------------------------------------------------------
// Pre-pass 2: convert W2 to f16, padding the K-dim (DK=240) to 256 with zeros
// so stage 2 runs 8 clean K=32 WMMA steps.
// w2c layout: [k][l][ddp]  (K-dim dd contiguous)
// ---------------------------------------------------------------------------
__global__ void convert_w2_kernel(const float* __restrict__ W2,
                                  _Float16* __restrict__ w2c) {
    int o = blockIdx.x * blockDim.x + threadIdx.x;
    if (o >= KEXP * L * DKP) return;
    int k  = o / (L * DKP);
    int r  = o % (L * DKP);
    int l  = r / DKP;
    int dd = r % DKP;
    float v = (dd < DK) ? W2[((size_t)(k * L + l)) * DK + dd] : 0.0f;
    w2c[o] = (_Float16)v;
}

// ---------------------------------------------------------------------------
// Main fused kernel: one workgroup = (64-row batch tile) x (one expert).
// 256 threads = 8 waves (wave32).
// ---------------------------------------------------------------------------
__launch_bounds__(256, 2)
__global__ void comp_mlp_kernel(const float*    __restrict__ x,
                                const _Float16* __restrict__ w1c,
                                const float*    __restrict__ b1,
                                const _Float16* __restrict__ w2c,
                                const float*    __restrict__ b2,
                                float*          __restrict__ out) {
    __shared__ __align__(16) _Float16 x_lds[BT][M];    // 16 KB
    __shared__ __align__(16) _Float16 h_lds[BT][DKP];  // 32 KB

    const int tid    = threadIdx.x;
    const int lane   = tid & 31;
    const int wave   = tid >> 5;
    const int laneLo = lane & 15;   // N (or M) index within a 16-wide tile
    const int hi     = lane >> 4;   // lane half selector for A/B K packing

    const int blk     = blockIdx.x;
    const int k       = blk & (KEXP - 1);
    const int rowBase = (blk >> 4) * BT;   // batch row base of this tile

    // ---- Stage x-tile into LDS with inline f32 -> f16 conversion ----------
    {
        const float4* src = (const float4*)(x + (size_t)rowBase * NCOL + (size_t)k * M);
        #pragma unroll
        for (int i = 0; i < 8; ++i) {
            int v    = tid + i * 256;        // 0..2047 float4 chunks
            int row  = v >> 5;               // 128 floats per row = 32 float4
            int col4 = v & 31;
            float4 d = src[(size_t)row * (NCOL / 4) + col4];
            half4 h4;
            h4[0] = (_Float16)d.x;
            h4[1] = (_Float16)d.y;
            h4[2] = (_Float16)d.z;
            h4[3] = (_Float16)d.w;
            *(half4*)&x_lds[row][col4 * 4] = h4;   // one ds_store_b64
        }
        // Zero the h padding region (cols 240..255): 64*16 = 1024 halfs.
        int prow = tid >> 2;
        int pcol = DK + (tid & 3) * 4;
        *(half4*)&h_lds[prow][pcol] = (half4){(_Float16)0, (_Float16)0,
                                              (_Float16)0, (_Float16)0};
    }
    __syncthreads();

    // ---- Stage 1: h = leaky(x @ W1b^T + b1), 4x15 = 60 (16x16) tiles ------
    for (int t = wave; t < 60; t += 8) {
        const int m0 = (t / 15) * 16;   // batch-row tile
        const int d0 = (t % 15) * 16;   // hidden-col tile
        // Bias folded into accumulator init: C/D column == laneLo for every
        // accumulator VGPR, so a single per-lane scalar seeds all 8.
        const float bias = b1[k * DK + d0 + laneLo];
        v8f acc = {bias, bias, bias, bias, bias, bias, bias, bias};
        #pragma unroll
        for (int kk = 0; kk < 4; ++kk) {          // K = 128 in steps of 32
            const int k0 = kk * 32;
            // A fragment (16x32 f16, contiguous K order k = 16*hi + s) from LDS
            const _Float16* ap = &x_lds[m0 + laneLo][0];
            half8 a0 = *(const half8*)(ap + k0 + hi * 16);
            half8 a1 = *(const half8*)(ap + k0 + hi * 16 + 8);
            // B fragment (32x16 f16): row = output col d, K contiguous, from L2
            const _Float16* bp =
                w1c + ((size_t)(k * DK + d0 + laneLo)) * M + k0 + hi * 16;
            half8 bv0 = *(const half8*)bp;
            half8 bv1 = *(const half8*)(bp + 8);
            acc = __builtin_amdgcn_wmma_f32_16x16x32_f16(
                false, cat16(a0, a1), false, cat16(bv0, bv1),
                (short)0, acc, false, false);
        }
        #pragma unroll
        for (int r = 0; r < 8; ++r) {   // C/D layout: VGPR r -> M = r + hi*8
            h_lds[m0 + r + hi * 8][d0 + laneLo] = (_Float16)leaky(acc[r]);
        }
    }
    __syncthreads();

    // ---- Stage 2: o = leaky(h @ W2^T + b2), 4x8 = 32 (16x16) tiles --------
    for (int t = wave; t < 32; t += 8) {
        const int m0 = (t >> 3) * 16;   // batch-row tile
        const int n0 = (t & 7) * 16;    // output-col tile
        const float bias = b2[k * L + n0 + laneLo];
        v8f acc = {bias, bias, bias, bias, bias, bias, bias, bias};
        #pragma unroll
        for (int kk = 0; kk < 8; ++kk) {          // K = 256 (padded) steps of 32
            const int d0 = kk * 32;
            const _Float16* ap = &h_lds[m0 + laneLo][0];
            half8 a0 = *(const half8*)(ap + d0 + hi * 16);
            half8 a1 = *(const half8*)(ap + d0 + hi * 16 + 8);
            const _Float16* bp =
                w2c + ((size_t)(k * L + n0 + laneLo)) * DKP + d0 + hi * 16;
            half8 bv0 = *(const half8*)bp;
            half8 bv1 = *(const half8*)(bp + 8);
            acc = __builtin_amdgcn_wmma_f32_16x16x32_f16(
                false, cat16(a0, a1), false, cat16(bv0, bv1),
                (short)0, acc, false, false);
        }
        float* orow = out + (size_t)(rowBase + m0) * NCOL + (size_t)k * L + n0 + laneLo;
        #pragma unroll
        for (int r = 0; r < 8; ++r) {
            orow[(size_t)(r + hi * 8) * NCOL] = leaky(acc[r]);
        }
    }
}

// ---------------------------------------------------------------------------
// Launch
// ---------------------------------------------------------------------------
extern "C" void kernel_launch(void* const* d_in, const int* in_sizes, int n_in,
                              void* d_out, int out_size, void* d_ws, size_t ws_size,
                              hipStream_t stream) {
    const float* x  = (const float*)d_in[0];
    const float* W1 = (const float*)d_in[1];
    const float* b1 = (const float*)d_in[2];
    const float* W2 = (const float*)d_in[3];
    const float* b2 = (const float*)d_in[4];
    float* out = (float*)d_out;

    // Workspace layout: f16 W1 diagonal blocks, then f16 padded W2 (~2 MB).
    _Float16* w1c = (_Float16*)d_ws;
    _Float16* w2c = (_Float16*)((char*)d_ws + (size_t)KEXP * DK * M * sizeof(_Float16));

    const int nW1 = KEXP * DK * M;    // 491520
    const int nW2 = KEXP * L * DKP;   // 524288
    convert_w1_kernel<<<(nW1 + 255) / 256, 256, 0, stream>>>(W1, w1c);
    convert_w2_kernel<<<(nW2 + 255) / 256, 256, 0, stream>>>(W2, w2c);

    const int Bv = out_size / (KEXP * L);   // batch size (16384)
    comp_mlp_kernel<<<(Bv / BT) * KEXP, 256, 0, stream>>>(x, w1c, b1, w2c, b2, out);
}
